// StructuredSU2Model_6880537608280
// MI455X (gfx1250) — matrix operand order repaired and checked by reading, not compiled, not run on hardware
//
#include <hip/hip_runtime.h>
#include <math.h>

typedef __attribute__((ext_vector_type(16))) _Float16 v16h;
typedef __attribute__((ext_vector_type(8)))  float    v8f;

#define LSZ     1024
#define NROWS   (LSZ * LSZ * 2)          // 2,097,152 MLP rows
#define NTILES  (NROWS / 16)             // 131,072 16-row WMMA tiles
#define U_FLOATS (NROWS * 8)             // U output: 8 floats (4 complex) per row

// Hardware tanh (CDNA5 V_TANH_F32 transcendental) if the builtin exists,
// else fall back to libm expansion.
#if defined(__has_builtin)
#if __has_builtin(__builtin_amdgcn_tanhf)
#define FAST_TANH(x) __builtin_amdgcn_tanhf(x)
#elif __has_builtin(__builtin_amdgcn_tanh_f32)
#define FAST_TANH(x) __builtin_amdgcn_tanh_f32(x)
#endif
#endif
#ifndef FAST_TANH
#define FAST_TANH(x) tanhf(x)
#endif

// Build a 16x32 f16 A-fragment from 32 contiguous f32s (one embedding pair half).
// ISA layout (CDNA5 ISA): lane half 'hi': halfs 0..7 = K(8hi..8hi+7),
// halfs 8..15 = K(16+8hi .. 16+8hi+7).
__device__ __forceinline__ v16h load_a_frag_f32(const float* __restrict__ p, int hi) {
    float4 f0 = *(const float4*)(p + hi * 8);
    float4 f1 = *(const float4*)(p + hi * 8 + 4);
    float4 f2 = *(const float4*)(p + 16 + hi * 8);
    float4 f3 = *(const float4*)(p + 16 + hi * 8 + 4);
    v16h a;
    a[0]  = (_Float16)f0.x; a[1]  = (_Float16)f0.y; a[2]  = (_Float16)f0.z; a[3]  = (_Float16)f0.w;
    a[4]  = (_Float16)f1.x; a[5]  = (_Float16)f1.y; a[6]  = (_Float16)f1.z; a[7]  = (_Float16)f1.w;
    a[8]  = (_Float16)f2.x; a[9]  = (_Float16)f2.y; a[10] = (_Float16)f2.z; a[11] = (_Float16)f2.w;
    a[12] = (_Float16)f3.x; a[13] = (_Float16)f3.y; a[14] = (_Float16)f3.z; a[15] = (_Float16)f3.w;
    return a;
}

__global__ __launch_bounds__(256) void su2_mlp_wmma_kernel(
    const float* __restrict__ node, const float* __restrict__ W1,
    const float* __restrict__ b1,   const float* __restrict__ W2,
    const float* __restrict__ b2,   const float* __restrict__ W3,
    const float* __restrict__ b3,   float* __restrict__ out)
{
    __shared__ float smem[8][16 * 64];   // per-wave 16x64 staging (f16 layer1 / f32 layer2)
    __shared__ float w3s[3][64];         // W3 transposed: w3s[c][k]
    __shared__ float b3s[3];

    const int tid  = threadIdx.x;
    const int lane = tid & 31;
    const int wv   = tid >> 5;
    const int lrow = lane & 15;          // tile row owned by this lane (A layout)
    const int hi   = lane >> 4;          // lane half -> K-sub-block select
    const int colb = lane & 15;          // column owned by this lane (B/C layout)

    if (tid < 64 * 3) w3s[tid % 3][tid / 3] = W3[tid];   // W3 is (64,3) row-major
    if (tid < 3)      b3s[tid] = b3[tid];
    __syncthreads();

    // ---- Preload all W1/W2 B-fragments into registers (loaded once per wave).
    // B 32x16 f16 layout: lane half 'hi': half e<8 = K(8*hi + e), e>=8 = K(16 + 8*hi + e-8), column = lane&15.
    v16h B1f[2][4], B2f[2][4];
    #pragma unroll
    for (int kc = 0; kc < 2; ++kc) {
        #pragma unroll
        for (int nt = 0; nt < 4; ++nt) {
            v16h f1, f2;
            const int kb  = kc * 32 + hi * 8;
            const int col = nt * 16 + colb;
            #pragma unroll
            for (int e = 0; e < 16; ++e) {
                f1[e] = (_Float16)W1[(kb + e + (e >= 8 ? 8 : 0)) * 64 + col];
                f2[e] = (_Float16)W2[(kb + e + (e >= 8 ? 8 : 0)) * 64 + col];
            }
            B1f[kc][nt] = f1;
            B2f[kc][nt] = f2;
        }
    }
    float bv1[4], bv2[4];
    #pragma unroll
    for (int nt = 0; nt < 4; ++nt) {
        bv1[nt] = b1[nt * 16 + colb];
        bv2[nt] = b2[nt * 16 + colb];
    }

    _Float16* zh = (_Float16*)&smem[wv][0];
    float*    zf = &smem[wv][0];

    const int wave_id = blockIdx.x * 8 + wv;
    const int nwaves  = gridDim.x * 8;

    for (int t = wave_id; t < NTILES; t += nwaves) {
        // ---- decode this lane's row: r = (x*L + y)*2 + m
        const int r  = t * 16 + lrow;
        const int x  = r >> 11;
        const int y  = (r >> 1) & (LSZ - 1);
        const int m  = r & 1;
        const int xn = (m == 0) ? ((x + 1) & (LSZ - 1)) : x;
        const int yn = (m == 0) ? y : ((y + 1) & (LSZ - 1));

        const float* ps = node + (size_t)(((x  << 10) | y ) << 5);
        const float* pn = node + (size_t)(((xn << 10) | yn) << 5);

        // input K-chunks: K[0..31] = h_self, K[32..63] = h_neighbor
        v16h A0 = load_a_frag_f32(ps, hi);
        v16h A1 = load_a_frag_f32(pn, hi);

        // ---- layer 1: z1 = tanh(inp @ W1 + b1)  (8 WMMA)
        v8f C[4];
        #pragma unroll
        for (int nt = 0; nt < 4; ++nt) {
            v8f c;
            #pragma unroll
            for (int e = 0; e < 8; ++e) c[e] = bv1[nt];
            c = __builtin_amdgcn_wmma_f32_16x16x32_f16(false, A0, false, B1f[0][nt],
                                                       (short)0, c, false, false);
            c = __builtin_amdgcn_wmma_f32_16x16x32_f16(false, A1, false, B1f[1][nt],
                                                       (short)0, c, false, false);
            C[nt] = c;
        }
        // C layout -> LDS f16 (row = v + 8*hi, col = 16*nt + colb)
        #pragma unroll
        for (int nt = 0; nt < 4; ++nt)
            #pragma unroll
            for (int v = 0; v < 8; ++v)
                zh[(v + hi * 8) * 64 + nt * 16 + colb] = (_Float16)FAST_TANH(C[nt][v]);
        asm volatile("" ::: "memory");   // same-wave LDS is in-order; fence the compiler

        // ---- re-read z1 in A-fragment layout
        v16h Z0, Z1;
        {
            const int base0 = lrow * 64 + hi * 8;         // kc=0
            uint4 lo = *(const uint4*)(zh + base0);
            uint4 up = *(const uint4*)(zh + base0 + 16);
            *((uint4*)&Z0)     = lo;
            *((uint4*)&Z0 + 1) = up;
            const int base1 = lrow * 64 + 32 + hi * 8;    // kc=1
            lo = *(const uint4*)(zh + base1);
            up = *(const uint4*)(zh + base1 + 16);
            *((uint4*)&Z1)     = lo;
            *((uint4*)&Z1 + 1) = up;
        }
        asm volatile("" ::: "memory");

        // ---- layer 2: z2 = tanh(z1 @ W2 + b2)  (8 WMMA)
        #pragma unroll
        for (int nt = 0; nt < 4; ++nt) {
            v8f c;
            #pragma unroll
            for (int e = 0; e < 8; ++e) c[e] = bv2[nt];
            c = __builtin_amdgcn_wmma_f32_16x16x32_f16(false, Z0, false, B2f[0][nt],
                                                       (short)0, c, false, false);
            c = __builtin_amdgcn_wmma_f32_16x16x32_f16(false, Z1, false, B2f[1][nt],
                                                       (short)0, c, false, false);
            C[nt] = c;
        }
        #pragma unroll
        for (int nt = 0; nt < 4; ++nt)
            #pragma unroll
            for (int v = 0; v < 8; ++v)
                zf[(v + hi * 8) * 64 + nt * 16 + colb] = FAST_TANH(C[nt][v]);
        asm volatile("" ::: "memory");

        // ---- layer 3: a = z2 @ W3 + b3  (VALU; K split across lane halves)
        float s0 = 0.f, s1 = 0.f, s2 = 0.f;
        {
            const int kb = hi * 32;
            const float* zr  = &zf[lrow * 64 + kb];
            const float* w0p = &w3s[0][kb];
            const float* w1p = &w3s[1][kb];
            const float* w2p = &w3s[2][kb];
            #pragma unroll 8
            for (int j = 0; j < 32; ++j) {
                const float z = zr[j];
                s0 = fmaf(z, w0p[j], s0);
                s1 = fmaf(z, w1p[j], s1);
                s2 = fmaf(z, w2p[j], s2);
            }
        }
        s0 += __shfl_xor(s0, 16, 32);
        s1 += __shfl_xor(s1, 16, 32);
        s2 += __shfl_xor(s2, 16, 32);

        const float a0 = s0 + b3s[0];
        const float a1 = s1 + b3s[1];
        const float a2 = s2 + b3s[2];

        // ---- su2_exp: U = cos|a| I + i sin|a|/|a| (a.sigma)
        const float th  = __builtin_sqrtf(a0 * a0 + a1 * a1 + a2 * a2);
        const float cth = __cosf(th);                       // hardware v_cos_f32 path
        const float s   = (th > 1e-8f) ? (__sinf(th) / th)  // hardware v_sin_f32 path
                                       : (1.0f - th * th * (1.0f / 6.0f));
        const float sa1 = s * a0, sa2 = s * a1, sa3 = s * a2;

        if (hi == 0) {
            // U row: [u00.re, u00.im, u01.re, u01.im, u10.re, u10.im, u11.re, u11.im]
            float4 u_lo = { cth,  sa3,  sa2,  sa1 };
            float4 u_hi = { -sa2, sa1,  cth, -sa3 };
            float4* up = (float4*)(out + (size_t)r * 8);
            up[0] = u_lo;
            up[1] = u_hi;
        } else {
            float* ap = out + (size_t)U_FLOATS + (size_t)r * 3;
            ap[0] = a0; ap[1] = a1; ap[2] = a2;
        }
    }
}

extern "C" void kernel_launch(void* const* d_in, const int* in_sizes, int n_in,
                              void* d_out, int out_size, void* d_ws, size_t ws_size,
                              hipStream_t stream) {
    (void)in_sizes; (void)n_in; (void)out_size; (void)d_ws; (void)ws_size;
    const float* node = (const float*)d_in[0];
    const float* W1   = (const float*)d_in[1];
    const float* b1   = (const float*)d_in[2];
    const float* W2   = (const float*)d_in[3];
    const float* b2   = (const float*)d_in[4];
    const float* W3   = (const float*)d_in[5];
    const float* b3   = (const float*)d_in[6];
    float* out = (float*)d_out;

    dim3 grid(1024);   // 8192 waves; 131072 tiles -> 16 tiles/wave, uniform
    dim3 block(256);   // 8 waves/block (wave32)
    hipLaunchKernelGGL(su2_mlp_wmma_kernel, grid, block, 0, stream,
                       node, W1, b1, W2, b2, W3, b3, out);
}
